// Attention_38749194945188
// MI455X (gfx1250) — matrix operand order repaired and checked by reading, not compiled, not run on hardware
//
#include <hip/hip_runtime.h>
#include <hip/hip_bf16.h>
#include <stdint.h>

// ---------------------------------------------------------------------------
// Attention (LN + QKV + masked softmax + out-proj) for gfx1250, bf16 WMMA path.
// B=2, N=2048, DIM=1024, HEADS=16, DHEAD=64.
// ---------------------------------------------------------------------------

#define DIM      1024
#define HEADS    16
#define DHEAD    64
#define INNER    1024
#define BATCH    2
#define SEQ      2048
#define ROWS     (BATCH * SEQ)        // 4096
#define QKV_N    (3 * INNER)          // 3072
#define SCALE    0.125f               // 64^-0.5
#define LN_EPS   1e-5f

typedef __bf16 bf16;
typedef bf16  v16bf __attribute__((ext_vector_type(16)));
typedef bf16  v8bf  __attribute__((ext_vector_type(8)));
typedef float v8f   __attribute__((ext_vector_type(8)));

#define V8F_ZERO {0.f,0.f,0.f,0.f,0.f,0.f,0.f,0.f}

// ------------------------- fragment helpers --------------------------------
// A fragment (16x32, bf16): lane m = lane&15, half = lane>>4.
// element e -> k = 16*(e/8) + 8*half + (e%8)  => two contiguous 8-elem runs.
__device__ __forceinline__ v16bf load_afrag(const bf16* __restrict__ base,
                                            int stride, int row0, int k0, int lane) {
    const int m = lane & 15, half = lane >> 4;
    const bf16* p = base + (size_t)(row0 + m) * stride + k0 + 8 * half;
    union { v16bf v; v8bf h[2]; } u;
    u.h[0] = *(const v8bf*)(p);
    u.h[1] = *(const v8bf*)(p + 16);
    return u.v;
}

// B fragment (32x16, bf16): lane n = lane&15, half = lane>>4,
// element e -> k = 16*(e/8) + 8*half + (e%8)  => two contiguous 8-elem runs.
__device__ __forceinline__ v16bf load_bfrag(const bf16* __restrict__ base,
                                            int stride, int n0, int k0, int lane) {
    const int n = lane & 15, half = lane >> 4;
    const bf16* p = base + (size_t)(n0 + n) * stride + k0 + 8 * half;
    union { v16bf v; v8bf h[2]; } u;
    u.h[0] = *(const v8bf*)(p);
    u.h[1] = *(const v8bf*)(p + 16);
    return u.v;
}

__device__ __forceinline__ v8f wmma_bf16(v16bf a, v16bf b, v8f c) {
    return __builtin_amdgcn_wmma_f32_16x16x32_bf16(
        false, a, false, b, (short)0, c, false, false);
}

// order-preserving float<->uint mapping for atomicMin
__device__ __forceinline__ unsigned int fenc(float f) {
    unsigned int u = __float_as_uint(f);
    return (u & 0x80000000u) ? ~u : (u | 0x80000000u);
}
__device__ __forceinline__ float fdec(unsigned int v) {
    unsigned int u = (v & 0x80000000u) ? (v & 0x7FFFFFFFu) : ~v;
    return __uint_as_float(u);
}

// ------------------------- kernel 1: LayerNorm + cast ----------------------
__global__ void ln_cast_kernel(const float* __restrict__ x,
                               const float* __restrict__ gamma,
                               const float* __restrict__ beta,
                               bf16* __restrict__ xn) {
    const int row = blockIdx.x;           // 0..4095
    const int tid = threadIdx.x;          // 256 threads
    const float* xr = x + (size_t)row * DIM;
    float v[4];
    float s = 0.f;
#pragma unroll
    for (int i = 0; i < 4; i++) { v[i] = xr[tid + 256 * i]; s += v[i]; }
    __shared__ float red[256];
    red[tid] = s; __syncthreads();
    for (int off = 128; off > 0; off >>= 1) {
        if (tid < off) red[tid] += red[tid + off];
        __syncthreads();
    }
    const float mu = red[0] * (1.f / DIM);
    __syncthreads();
    float sq = 0.f;
#pragma unroll
    for (int i = 0; i < 4; i++) { float d = v[i] - mu; sq += d * d; }
    red[tid] = sq; __syncthreads();
    for (int off = 128; off > 0; off >>= 1) {
        if (tid < off) red[tid] += red[tid + off];
        __syncthreads();
    }
    const float rstd = rsqrtf(red[0] * (1.f / DIM) + LN_EPS);
#pragma unroll
    for (int i = 0; i < 4; i++) {
        int c = tid + 256 * i;
        xn[(size_t)row * DIM + c] = (bf16)((v[i] - mu) * rstd * gamma[c] + beta[c]);
    }
}

// ------------------------- kernel 2: cast+transpose weights ----------------
__global__ void cvt_weights_kernel(const float* __restrict__ w_qkv,
                                   const float* __restrict__ w_out,
                                   bf16* __restrict__ wqkvT,   // [3072][1024]
                                   bf16* __restrict__ woutT) { // [1024][1024]
    const int total1 = QKV_N * DIM;
    int idx = blockIdx.x * blockDim.x + threadIdx.x;
    if (idx < total1) {
        int n = idx / DIM, k = idx % DIM;
        wqkvT[idx] = (bf16)w_qkv[(size_t)k * QKV_N + n];
    } else {
        int j = idx - total1;
        if (j < DIM * DIM) {
            int n = j / DIM, k = j % DIM;
            woutT[j] = (bf16)w_out[(size_t)k * DIM + n];
        }
    }
}

__global__ void init_min_kernel(unsigned int* minbuf) { minbuf[0] = 0xFFFFFFFFu; }

// ------------------------- kernel 3: QKV GEMM ------------------------------
// C[4096 x 3072] = Xn @ WqkvT^T ; one wave per 32x64 tile (2 A frags reuse
// each B frag; 8 WMMA per 12 b128 loads in the k-loop).
// Writes Q(b,h,n,d)*SCALE, K(b,h,n,d), V transposed (b,h,d,n) as bf16.
__global__ void qkv_gemm_kernel(const bf16* __restrict__ xn,
                                const bf16* __restrict__ wT,
                                bf16* __restrict__ Q,
                                bf16* __restrict__ K,
                                bf16* __restrict__ Vt) {
    const int lane = threadIdx.x & 31;
    const int wave = blockIdx.x * (blockDim.x >> 5) + (threadIdx.x >> 5);
    const int rowtile = wave / (QKV_N / 64);   // 0..127 (32-row tiles)
    const int coltile = wave % (QKV_N / 64);   // 0..47
    const int m0 = rowtile * 32, n0 = coltile * 64;

    v8f acc[2][4] = {{V8F_ZERO, V8F_ZERO, V8F_ZERO, V8F_ZERO},
                     {V8F_ZERO, V8F_ZERO, V8F_ZERO, V8F_ZERO}};
    for (int k0 = 0; k0 < DIM; k0 += 32) {
        v16bf a0 = load_afrag(xn, DIM, m0, k0, lane);
        v16bf a1 = load_afrag(xn, DIM, m0 + 16, k0, lane);
#pragma unroll
        for (int t = 0; t < 4; t++) {
            v16bf b = load_bfrag(wT, DIM, n0 + t * 16, k0, lane);
            acc[0][t] = wmma_bf16(a0, b, acc[0][t]);
            acc[1][t] = wmma_bf16(a1, b, acc[1][t]);
        }
    }
    const int half = lane >> 4, ln = lane & 15;
    const int which = n0 / INNER;             // 0=Q 1=K 2=V (uniform per wave)
    const int h = (n0 % INNER) / DHEAD;       // uniform per wave
#pragma unroll
    for (int g = 0; g < 2; g++) {
#pragma unroll
        for (int t = 0; t < 4; t++) {
            const int dd = t * 16 + ln;       // 0..63
#pragma unroll
            for (int r = 0; r < 8; r++) {
                const int row = m0 + g * 16 + r + 8 * half;
                const int bb = row >> 11, nn = row & 2047;
                const size_t head = (size_t)(bb * HEADS + h);
                const float v = acc[g][t][r];
                if (which == 0)
                    Q[(head * SEQ + nn) * DHEAD + dd] = (bf16)(v * SCALE);
                else if (which == 1)
                    K[(head * SEQ + nn) * DHEAD + dd] = (bf16)v;
                else
                    Vt[(head * DHEAD + dd) * SEQ + nn] = (bf16)v;
            }
        }
    }
}

// ------------------------- kernel 4: global min of dots --------------------
// One wave per (b,h,16-row tile): recompute S = Qs K^T, reduce min, atomicMin.
__global__ void smin_kernel(const bf16* __restrict__ Q,
                            const bf16* __restrict__ K,
                            unsigned int* __restrict__ minbuf) {
    const int lane = threadIdx.x & 31;
    const int wave = blockIdx.x * (blockDim.x >> 5) + (threadIdx.x >> 5);
    const int bb = wave / (HEADS * (SEQ / 16));
    const int rem = wave % (HEADS * (SEQ / 16));
    const int h = rem / (SEQ / 16);
    const int it = rem % (SEQ / 16);
    const int i0 = it * 16;
    const bf16* Qh = Q + (size_t)(bb * HEADS + h) * SEQ * DHEAD;
    const bf16* Kh = K + (size_t)(bb * HEADS + h) * SEQ * DHEAD;

    const v16bf aq0 = load_afrag(Qh, DHEAD, i0, 0, lane);
    const v16bf aq1 = load_afrag(Qh, DHEAD, i0, 32, lane);
    float wmin = __builtin_inff();
    for (int j0 = 0; j0 < SEQ; j0 += 16) {
        v8f s = V8F_ZERO;
        s = wmma_bf16(aq0, load_bfrag(Kh, DHEAD, j0, 0, lane), s);
        s = wmma_bf16(aq1, load_bfrag(Kh, DHEAD, j0, 32, lane), s);
#pragma unroll
        for (int r = 0; r < 8; r++) wmin = fminf(wmin, s[r]);
    }
    for (int xm = 16; xm > 0; xm >>= 1)
        wmin = fminf(wmin, __shfl_xor(wmin, xm, 32));
    if (lane == 0) atomicMin(minbuf, fenc(wmin));
}

// ------------------------- kernel 5: flash attention -----------------------
// One wave per (b,h,16-row tile). Online softmax over j in steps of 32.
__global__ void attn_kernel(const bf16* __restrict__ Q,
                            const bf16* __restrict__ K,
                            const bf16* __restrict__ Vt,
                            const unsigned char* __restrict__ mask, // bool[b,n,n]
                            const unsigned int* __restrict__ minbuf,
                            bf16* __restrict__ outh) {              // [4096][1024]
    __shared__ __align__(16) bf16 plds[8][16 * 32];
    const int wid = threadIdx.x >> 5, lane = threadIdx.x & 31;
    const int wave = blockIdx.x * 8 + wid;
    const int bb = wave / (HEADS * (SEQ / 16));
    const int rem = wave % (HEADS * (SEQ / 16));
    const int h = rem / (SEQ / 16);
    const int it = rem % (SEQ / 16);
    const int i0 = it * 16;
    const int half = lane >> 4, ln = lane & 15;

    const float min_val = fdec(minbuf[0]);
    const bf16* Qh = Q + (size_t)(bb * HEADS + h) * SEQ * DHEAD;
    const bf16* Kh = K + (size_t)(bb * HEADS + h) * SEQ * DHEAD;
    const bf16* Vh = Vt + (size_t)(bb * HEADS + h) * DHEAD * SEQ;
    const unsigned char* mb = mask + (size_t)bb * SEQ * SEQ;
    bf16* pl = plds[wid];

    const v16bf aq0 = load_afrag(Qh, DHEAD, i0, 0, lane);
    const v16bf aq1 = load_afrag(Qh, DHEAD, i0, 32, lane);

    v8f o[4] = {V8F_ZERO, V8F_ZERO, V8F_ZERO, V8F_ZERO};
    float mrun[8], lrun[8];
#pragma unroll
    for (int r = 0; r < 8; r++) { mrun[r] = -__builtin_inff(); lrun[r] = 0.f; }

    for (int j0 = 0; j0 < SEQ; j0 += 32) {
        // S tiles (already scaled: Q was pre-multiplied by SCALE)
        v8f s0 = V8F_ZERO, s1 = V8F_ZERO;
        s0 = wmma_bf16(aq0, load_bfrag(Kh, DHEAD, j0,      0,  lane), s0);
        s0 = wmma_bf16(aq1, load_bfrag(Kh, DHEAD, j0,      32, lane), s0);
        s1 = wmma_bf16(aq0, load_bfrag(Kh, DHEAD, j0 + 16, 0,  lane), s1);
        s1 = wmma_bf16(aq1, load_bfrag(Kh, DHEAD, j0 + 16, 32, lane), s1);

        float p0[8], p1[8];
#pragma unroll
        for (int r = 0; r < 8; r++) {
            const int irow = i0 + r + 8 * half;
            const unsigned char* mp = mb + (size_t)irow * SEQ + j0;
            const float v0 = mp[ln]      ? s0[r] : min_val;
            const float v1 = mp[16 + ln] ? s1[r] : min_val;
            float tm = fmaxf(v0, v1);
            for (int xm = 1; xm < 16; xm <<= 1)
                tm = fmaxf(tm, __shfl_xor(tm, xm, 32));
            const float mnew = fmaxf(mrun[r], tm);
            const float corr = __expf(mrun[r] - mnew);
            const float e0 = __expf(v0 - mnew);
            const float e1 = __expf(v1 - mnew);
            float rs = e0 + e1;
            for (int xm = 1; xm < 16; xm <<= 1)
                rs += __shfl_xor(rs, xm, 32);
            lrun[r] = lrun[r] * corr + rs;
            mrun[r] = mnew;
            p0[r] = e0; p1[r] = e1;
#pragma unroll
            for (int t = 0; t < 4; t++) o[t][r] *= corr;
        }
        // C-layout -> A-fragment reshape of P through LDS (row-major 16x32)
#pragma unroll
        for (int r = 0; r < 8; r++) {
            const int row = r + 8 * half;
            pl[row * 32 + ln]      = (bf16)p0[r];
            pl[row * 32 + 16 + ln] = (bf16)p1[r];
        }
        asm volatile("s_wait_dscnt 0" ::: "memory");   // CDNA5 split DS counter
        union { v16bf v; v8bf hh[2]; } up;
        up.hh[0] = *(const v8bf*)(pl + ln * 32 + 8 * half);
        up.hh[1] = *(const v8bf*)(pl + ln * 32 + 16 + 8 * half);
        const v16bf pf = up.v;
#pragma unroll
        for (int t = 0; t < 4; t++) {
            v16bf bv = load_bfrag(Vh, SEQ, t * 16, j0, lane);  // V transposed
            o[t] = wmma_bf16(pf, bv, o[t]);
        }
    }
    // epilogue: divide by row sums, write out_heads bf16
#pragma unroll
    for (int t = 0; t < 4; t++) {
#pragma unroll
        for (int r = 0; r < 8; r++) {
            const int irow = i0 + r + 8 * half;
            const float v = o[t][r] / lrun[r];
            outh[(size_t)(bb * SEQ + irow) * INNER + h * DHEAD + t * 16 + ln] = (bf16)v;
        }
    }
}

// ------------------------- kernel 6: output projection ---------------------
// One wave per 32x64 tile, same B-reuse structure as QKV GEMM.
__global__ void out_gemm_kernel(const bf16* __restrict__ A,   // [4096][1024]
                                const bf16* __restrict__ wT,  // [1024][1024]
                                float* __restrict__ out) {    // [4096][1024]
    const int lane = threadIdx.x & 31;
    const int wave = blockIdx.x * (blockDim.x >> 5) + (threadIdx.x >> 5);
    const int rowtile = wave / (DIM / 64);   // 0..127 (32-row tiles)
    const int coltile = wave % (DIM / 64);   // 0..15
    const int m0 = rowtile * 32, n0 = coltile * 64;

    v8f acc[2][4] = {{V8F_ZERO, V8F_ZERO, V8F_ZERO, V8F_ZERO},
                     {V8F_ZERO, V8F_ZERO, V8F_ZERO, V8F_ZERO}};
    for (int k0 = 0; k0 < DIM; k0 += 32) {
        v16bf a0 = load_afrag(A, DIM, m0, k0, lane);
        v16bf a1 = load_afrag(A, DIM, m0 + 16, k0, lane);
#pragma unroll
        for (int t = 0; t < 4; t++) {
            v16bf b = load_bfrag(wT, DIM, n0 + t * 16, k0, lane);
            acc[0][t] = wmma_bf16(a0, b, acc[0][t]);
            acc[1][t] = wmma_bf16(a1, b, acc[1][t]);
        }
    }
    const int half = lane >> 4, ln = lane & 15;
#pragma unroll
    for (int g = 0; g < 2; g++)
#pragma unroll
        for (int t = 0; t < 4; t++)
#pragma unroll
            for (int r = 0; r < 8; r++)
                out[(size_t)(m0 + g * 16 + r + 8 * half) * DIM + n0 + t * 16 + ln]
                    = acc[g][t][r];
}

// ------------------------- launcher ----------------------------------------
extern "C" void kernel_launch(void* const* d_in, const int* in_sizes, int n_in,
                              void* d_out, int out_size, void* d_ws, size_t ws_size,
                              hipStream_t stream) {
    const float* x      = (const float*)d_in[0];
    const unsigned char* mask = (const unsigned char*)d_in[1]; // jax bool -> 1 byte/elem
    const float* gamma  = (const float*)d_in[2];
    const float* beta   = (const float*)d_in[3];
    const float* w_qkv  = (const float*)d_in[4];
    const float* w_out  = (const float*)d_in[5];
    float* out = (float*)d_out;

    // workspace layout (bf16 buffers), ~50.4 MB total
    char* ws = (char*)d_ws;
    bf16* xn    = (bf16*)(ws);                         // 4096*1024  (8 MB)
    bf16* wqkvT = (bf16*)(ws + 8388608);               // 3072*1024  (6 MB)
    bf16* woutT = (bf16*)(ws + 14680064);              // 1024*1024  (2 MB)
    bf16* Qb    = (bf16*)(ws + 16777216);              // 2*16*2048*64 (8 MB)
    bf16* Kb    = (bf16*)(ws + 25165824);              // 8 MB
    bf16* Vt    = (bf16*)(ws + 33554432);              // 8 MB
    bf16* outh  = (bf16*)(ws + 41943040);              // 8 MB
    unsigned int* minbuf = (unsigned int*)(ws + 50331648);

    ln_cast_kernel<<<ROWS, 256, 0, stream>>>(x, gamma, beta, xn);
    cvt_weights_kernel<<<(QKV_N * DIM + DIM * DIM + 255) / 256, 256, 0, stream>>>(
        w_qkv, w_out, wqkvT, woutT);
    init_min_kernel<<<1, 1, 0, stream>>>(minbuf);
    qkv_gemm_kernel<<<(128 * 48) / 8, 256, 0, stream>>>(xn, wqkvT, Qb, Kb, Vt);
    smin_kernel<<<(BATCH * HEADS * (SEQ / 16)) / 8, 256, 0, stream>>>(Qb, Kb, minbuf);
    attn_kernel<<<(BATCH * HEADS * (SEQ / 16)) / 8, 256, 0, stream>>>(
        Qb, Kb, Vt, mask, minbuf, outh);
    out_gemm_kernel<<<(128 * 16) / 8, 256, 0, stream>>>(outh, woutT, out);
}
